// fastGCN_71683004171209
// MI455X (gfx1250) — matrix operand order repaired and checked by reading, not compiled, not run on hardware
//
#include <hip/hip_runtime.h>
#include <hip/hip_bf16.h>

typedef __attribute__((ext_vector_type(16))) __bf16 v16bf;
typedef __attribute__((ext_vector_type(8)))  float  v8f;

// ---------------- degree / normalization ----------------
__global__ void k_deg_init(float* deg, int n) {
  int i = blockIdx.x * blockDim.x + threadIdx.x;
  if (i < n) deg[i] = 1.0f;  // self-loop contributes 1 to every node's degree
}

__global__ void k_deg_accum(const int* __restrict__ dst, float* deg, int e) {
  int i = blockIdx.x * blockDim.x + threadIdx.x;
  if (i < e) atomicAdd(&deg[dst[i]], 1.0f);
}

__global__ void k_dinv(float* deg, int n) {
  int i = blockIdx.x * blockDim.x + threadIdx.x;
  if (i < n) deg[i] = rsqrtf(deg[i]);  // deg >= 1 always
}

// ---------------- GEMM1: Y1[N,64] = bf16(x[N,128]) @ bf16(W1[128,64]) ----------------
// One wave computes a 16x64 output tile (4 accumulators, 4 K-steps -> 16 WMMAs).
__global__ void k_gemm1(const float* __restrict__ x, const float* __restrict__ W1,
                        float* __restrict__ Y1, int n) {
  // W1 transposed into LDS as Wt[n][k] (64 x 128 bf16 = 16KB) so each lane's
  // B-fragment (K = g*16 + j contiguous, N fixed) is 16 contiguous bf16.
  __shared__ __align__(32) __bf16 Wt[64 * 128];
  for (int idx = threadIdx.x; idx < 128 * 64; idx += blockDim.x) {
    int k = idx >> 6, c = idx & 63;
    Wt[c * 128 + k] = (__bf16)W1[idx];
  }
  __syncthreads();

  int wave = threadIdx.x >> 5;
  int lane = threadIdx.x & 31;
  int tile = blockIdx.x * 8 + wave;
  int ntiles = n >> 4;
  if (tile >= ntiles) return;

  int m0 = tile << 4;
  int g  = lane >> 4;       // lane group (0: lanes 0-15, 1: lanes 16-31)
  int ln = lane & 15;
  int rowA = m0 + ln;       // A-matrix row for this lane (M = lane%16)
  if (rowA >= n) rowA = n - 1;
  const float* xr = x + (size_t)rowA * 128;

  v8f acc[4] = {};
  #pragma unroll
  for (int kk = 0; kk < 4; ++kk) {
    int kb = kk * 32;
    // A fragment, contiguous K layout matching B: j=0..15 -> K=kb+g*16+j
    v16bf a;
    #pragma unroll
    for (int j = 0; j < 8; ++j) a[j]     = (__bf16)xr[kb + g * 16 + j];
    #pragma unroll
    for (int j = 0; j < 8; ++j) a[8 + j] = (__bf16)xr[kb + g * 16 + 8 + j];

    #pragma unroll
    for (int t = 0; t < 4; ++t) {
      // B fragment: N = t*16 + ln, K = kb + g*16 + j  (16 contiguous bf16 in Wt row)
      const v16bf* bp = reinterpret_cast<const v16bf*>(&Wt[(t * 16 + ln) * 128 + kb + g * 16]);
      v16bf b = *bp;
      acc[t] = __builtin_amdgcn_wmma_f32_16x16x32_bf16(
          false, a, false, b, (short)0, acc[t], false, false);
    }
  }

  // C/D layout: VGPR r -> row m0 + g*8 + r, col = n-tile*16 + ln
  #pragma unroll
  for (int t = 0; t < 4; ++t) {
    #pragma unroll
    for (int r = 0; r < 8; ++r) {
      int row = m0 + g * 8 + r;
      if (row < n) Y1[(size_t)row * 64 + t * 16 + ln] = acc[t][r];
    }
  }
}

// ---------------- self-loop term: H[i,c] = dinv[i]^2 * Y[i,c] ----------------
__global__ void k_selfinit(const float* __restrict__ Y, const float* __restrict__ dinv,
                           float* __restrict__ H, int total, int C) {
  int i = blockIdx.x * blockDim.x + threadIdx.x;
  if (i < total) {
    int r = i / C;
    float di = dinv[r];
    H[i] = di * di * Y[i];
  }
}

// ---------------- edge scatter, 64 channels: one wave per edge ----------------
__global__ void k_scatter64(const int* __restrict__ src, const int* __restrict__ dst,
                            const float* __restrict__ dinv, const float* __restrict__ Y,
                            float* H, int e) {
  int wave = (blockIdx.x * blockDim.x + threadIdx.x) >> 5;
  int lane = threadIdx.x & 31;
  if (wave >= e) return;
  int s = src[wave], d = dst[wave];
  float w = dinv[s] * dinv[d];
  const float* ys = Y + (size_t)s * 64;
  float* hd = H + (size_t)d * 64;
  atomicAdd(&hd[lane],      w * ys[lane]);
  atomicAdd(&hd[lane + 32], w * ys[lane + 32]);
}

// ---------------- edge scatter, 40 channels ----------------
__global__ void k_scatter40(const int* __restrict__ src, const int* __restrict__ dst,
                            const float* __restrict__ dinv, const float* __restrict__ Y,
                            float* O, int e) {
  int wave = (blockIdx.x * blockDim.x + threadIdx.x) >> 5;
  int lane = threadIdx.x & 31;
  if (wave >= e) return;
  int s = src[wave], d = dst[wave];
  float w = dinv[s] * dinv[d];
  const float* ys = Y + (size_t)s * 40;
  float* od = O + (size_t)d * 40;
  atomicAdd(&od[lane], w * ys[lane]);
  if (lane < 8) atomicAdd(&od[lane + 32], w * ys[lane + 32]);
}

// ---------------- GEMM2: Y2[N,40] = bf16(relu(H+b1)) @ bf16(W2[64,40]) ----------------
// K = 64 (2 K-steps), N padded to 48 (3 col-tiles), relu+bias fused into A-fragment load.
__global__ void k_gemm2(const float* __restrict__ H, const float* __restrict__ W2,
                        const float* __restrict__ b1, float* __restrict__ Y2, int n) {
  __shared__ __align__(32) __bf16 Wt[48 * 64];  // Wt[c][k], cols 40..47 zero-padded
  __shared__ float b1s[64];
  for (int idx = threadIdx.x; idx < 48 * 64; idx += blockDim.x) {
    int c = idx >> 6, k = idx & 63;
    Wt[idx] = (c < 40) ? (__bf16)W2[k * 40 + c] : (__bf16)0.0f;
  }
  if (threadIdx.x < 64) b1s[threadIdx.x] = b1[threadIdx.x];
  __syncthreads();

  int wave = threadIdx.x >> 5;
  int lane = threadIdx.x & 31;
  int tile = blockIdx.x * 8 + wave;
  int ntiles = n >> 4;
  if (tile >= ntiles) return;

  int m0 = tile << 4;
  int g  = lane >> 4;
  int ln = lane & 15;
  int rowA = m0 + ln;
  if (rowA >= n) rowA = n - 1;
  const float* hr = H + (size_t)rowA * 64;

  v8f acc[3] = {};
  #pragma unroll
  for (int kk = 0; kk < 2; ++kk) {
    int kb = kk * 32;
    v16bf a;
    #pragma unroll
    for (int j = 0; j < 8; ++j) {
      int k0 = kb + g * 16 + j;
      int k1 = kb + g * 16 + 8 + j;
      a[j]     = (__bf16)fmaxf(hr[k0] + b1s[k0], 0.0f);
      a[8 + j] = (__bf16)fmaxf(hr[k1] + b1s[k1], 0.0f);
    }
    #pragma unroll
    for (int t = 0; t < 3; ++t) {
      const v16bf* bp = reinterpret_cast<const v16bf*>(&Wt[(t * 16 + ln) * 64 + kb + g * 16]);
      v16bf b = *bp;
      acc[t] = __builtin_amdgcn_wmma_f32_16x16x32_bf16(
          false, a, false, b, (short)0, acc[t], false, false);
    }
  }

  #pragma unroll
  for (int t = 0; t < 3; ++t) {
    #pragma unroll
    for (int r = 0; r < 8; ++r) {
      int row = m0 + g * 8 + r;
      int col = t * 16 + ln;
      if (row < n && col < 40) Y2[(size_t)row * 40 + col] = acc[t][r];
    }
  }
}

// ---------------- log_softmax over 40 cols: one wave per row ----------------
__global__ void k_logsoftmax(const float* __restrict__ O, const float* __restrict__ b2,
                             float* __restrict__ out, int n) {
  int wave = (blockIdx.x * blockDim.x + threadIdx.x) >> 5;
  int lane = threadIdx.x & 31;
  if (wave >= n) return;
  const float* orow = O + (size_t)wave * 40;
  float v0 = orow[lane] + b2[lane];
  float v1 = (lane < 8) ? (orow[32 + lane] + b2[32 + lane]) : -3.0e38f;

  float m = fmaxf(v0, v1);
  #pragma unroll
  for (int off = 16; off; off >>= 1) m = fmaxf(m, __shfl_xor(m, off, 32));
  float s = __expf(v0 - m) + ((lane < 8) ? __expf(v1 - m) : 0.0f);
  #pragma unroll
  for (int off = 16; off; off >>= 1) s += __shfl_xor(s, off, 32);
  float lse = m + __logf(s);

  float* po = out + (size_t)wave * 40;
  po[lane] = v0 - lse;
  if (lane < 8) po[32 + lane] = v1 - lse;
}

// ---------------- host launcher ----------------
extern "C" void kernel_launch(void* const* d_in, const int* in_sizes, int n_in,
                              void* d_out, int out_size, void* d_ws, size_t ws_size,
                              hipStream_t stream) {
  const float* x  = (const float*)d_in[0];
  const int*   ei = (const int*)  d_in[1];
  const float* W1 = (const float*)d_in[2];
  const float* b1 = (const float*)d_in[3];
  const float* W2 = (const float*)d_in[4];
  const float* b2 = (const float*)d_in[5];
  float* out = (float*)d_out;

  const int N = in_sizes[0] / 128;   // 100000
  const int E = in_sizes[1] / 2;     // 800000
  const int* src = ei;
  const int* dst = ei + E;

  // Workspace: dinv | bufA (Y1 then Y2) | bufB (H then O)
  char* ws = (char*)d_ws;
  size_t degBytes = (((size_t)N * 4) + 255) & ~(size_t)255;
  size_t bufBytes = (((size_t)N * 64 * 4) + 255) & ~(size_t)255;
  float* dinv = (float*)ws;
  float* bufA = (float*)(ws + degBytes);
  float* bufB = (float*)(ws + degBytes + bufBytes);

  const int thr = 256;
  int nb  = (N + thr - 1) / thr;
  int eb  = (E + thr - 1) / thr;
  int ntiles = (N + 15) / 16;
  int gemmBlocks = (ntiles + 7) / 8;       // 8 waves / block
  int edgeWaveBlocks = (E + 7) / 8;        // 1 wave / edge
  int rowWaveBlocks = (N + 7) / 8;         // 1 wave / row

  // degrees & dinv
  k_deg_init<<<nb, thr, 0, stream>>>(dinv, N);
  k_deg_accum<<<eb, thr, 0, stream>>>(dst, dinv, E);
  k_dinv<<<nb, thr, 0, stream>>>(dinv, N);

  // layer 1: GEMM first (P commutes with channel GEMM), then propagate
  k_gemm1<<<gemmBlocks, thr, 0, stream>>>(x, W1, bufA, N);
  k_selfinit<<<((size_t)N * 64 + thr - 1) / thr, thr, 0, stream>>>(bufA, dinv, bufB, N * 64, 64);
  k_scatter64<<<edgeWaveBlocks, thr, 0, stream>>>(src, dst, dinv, bufA, bufB, E);

  // layer 2: fused relu+bias in GEMM A-load, then propagate
  k_gemm2<<<gemmBlocks, thr, 0, stream>>>(bufB, W2, b1, bufA, N);
  k_selfinit<<<((size_t)N * 40 + thr - 1) / thr, thr, 0, stream>>>(bufA, dinv, bufB, N * 40, 40);
  k_scatter40<<<edgeWaveBlocks, thr, 0, stream>>>(src, dst, dinv, bufA, bufB, E);

  // bias + log_softmax -> output
  k_logsoftmax<<<rowWaveBlocks, thr, 0, stream>>>(bufB, b2, out, N);
}
